// Laplacian_17351667876341
// MI455X (gfx1250) — compile-verified, matrix operand order checked by reading, not run on hardware
//
#include <hip/hip_runtime.h>

typedef __attribute__((ext_vector_type(16))) _Float16 v16h;
typedef __attribute__((ext_vector_type(8)))  float    v8f;

__device__ __forceinline__ float fast_tanh(float x) {
#if __has_builtin(__builtin_amdgcn_tanhf)
  return __builtin_amdgcn_tanhf(x);          // gfx1250 v_tanh_f32
#else
  return tanhf(x);
#endif
}

#define WAVES_PER_BLOCK 4
#define ROWS_PER_WAVE   16
#define NCHAN           5                    // h, y1[0..2], y2sum

// ---- LDS layout (bytes) ----
#define BFRAG_BYTES 16384                    // 16 frags * 2 groups * 32 lanes * 16B
#define W1_OFF      BFRAG_BYTES              // 128*3 f32 = 1536
#define B1_OFF      (W1_OFF + 1536)          // 128 f32
#define B2_OFF      (B1_OFF + 512)           // 64 f32
#define W3_OFF      (B2_OFF + 256)           // 64 f32
#define WAVE0_OFF   (W3_OFF + 256)           // = 18944
#define XS_BYTES    256                      // 16 rows * 3 f32, padded
#define CHAN_ROW_H  136                      // halfs/row: 128 + 8 pad (272B, 16B-aligned)
#define CHAN_H      (16 * CHAN_ROW_H)        // 2176 halfs per channel tile
#define WAVE_BYTES  (XS_BYTES + NCHAN * CHAN_H * 2)   // 256 + 21760 = 22016
#define SMEM_BYTES  (WAVE0_OFF + WAVES_PER_BLOCK * WAVE_BYTES)  // 107008

__global__ __launch_bounds__(WAVES_PER_BLOCK * 32)
void lap_mlp_kernel(const float* __restrict__ x,
                    const float* __restrict__ W1,
                    const float* __restrict__ b1,
                    const float* __restrict__ W2,
                    const float* __restrict__ b2,
                    const float* __restrict__ W3,
                    float* __restrict__ out,
                    int B)
{
  extern __shared__ char smem[];
  uint4*    bfrag4 = (uint4*)smem;
  _Float16* bh     = (_Float16*)smem;
  float* W1s = (float*)(smem + W1_OFF);
  float* b1s = (float*)(smem + B1_OFF);
  float* b2s = (float*)(smem + B2_OFF);
  float* w3s = (float*)(smem + W3_OFF);

  const int t    = threadIdx.x;
  const int lane = t & 31;
  const int wid  = t >> 5;

  char* wbase = smem + WAVE0_OFF + wid * WAVE_BYTES;
  float*    xs   = (float*)wbase;
  _Float16* chan = (_Float16*)(wbase + XS_BYTES);

  // ---------- cooperative staging: weights + W2 -> f16 B-fragment layout ----------
  for (int i = t; i < 384; i += WAVES_PER_BLOCK * 32) W1s[i] = W1[i];
  if (t < 128) b1s[t] = b1[t];
  if (t < 64)  { b2s[t] = b2[t]; w3s[t] = W3[t]; }

  // bfrag half index: ((fk*2+g)*32 + ln)*8 + s8 ; fk = n*4+kc, g selects VGPR0-3 / VGPR4-7
  for (int i = t; i < 8192; i += WAVES_PER_BLOCK * 32) {
    int s8 = i & 7;
    int ln = (i >> 3) & 31;
    int g  = (i >> 8) & 1;
    int fk = i >> 9;
    int slot = g * 8 + s8;
    int koff = (ln < 16) ? ((slot < 8) ? slot : slot + 8)
                         : ((slot < 8) ? slot + 8 : slot + 16);
    int n  = fk >> 2, kc = fk & 3;
    int k  = kc * 32 + koff;
    int nc = n * 16 + (ln & 15);
    bh[i] = (_Float16)W2[nc * 128 + k];      // B[k][n] = W2[n][k]
  }

  const int tileBase = (blockIdx.x * WAVES_PER_BLOCK + wid) * ROWS_PER_WAVE;

  for (int i = lane; i < 48; i += 32) {
    long gi = (long)tileBase * 3 + i;
    xs[i] = (gi < (long)B * 3) ? x[gi] : 0.0f;
  }
  __syncthreads();

  // ---------- layer 1 + tanh jet -> 5 f16 channel tiles [16 x 128] in LDS ----------
  // ch0: h = tanh(x·W1^T+b1)        (for z0 -> tanh'(z0) terms)
  // ch1..3: y1_d = s*W1[:,d]        (squared later -> keep separate)
  // ch4: y2sum = -2*h*s*Σ_d W1[:,d]^2   (linear into layer 2 -> pre-summed over d)
  const int f0 = lane * 4;
  #pragma unroll 4
  for (int row = 0; row < 16; ++row) {
    float xa = xs[row * 3 + 0], xb = xs[row * 3 + 1], xc = xs[row * 3 + 2];
    union { _Float16 h[4]; unsigned long long u; } v[NCHAN];
    #pragma unroll
    for (int j = 0; j < 4; ++j) {
      int f = f0 + j;
      float wa = W1s[f * 3 + 0], wb = W1s[f * 3 + 1], wc = W1s[f * 3 + 2];
      float pre = fmaf(xa, wa, fmaf(xb, wb, fmaf(xc, wc, b1s[f])));
      float h = fast_tanh(pre);
      float s = 1.0f - h * h;               // sech^2
      float m = -2.0f * h * s;
      v[0].h[j] = (_Float16)h;
      v[1].h[j] = (_Float16)(s * wa);
      v[2].h[j] = (_Float16)(s * wb);
      v[3].h[j] = (_Float16)(s * wc);
      v[4].h[j] = (_Float16)(m * (wa * wa + wb * wb + wc * wc));
    }
    #pragma unroll
    for (int c = 0; c < NCHAN; ++c)
      *(unsigned long long*)(chan + c * CHAN_H + row * CHAN_ROW_H + f0) = v[c].u;
  }
  __syncthreads();

  // ---------- load all A fragments (5 channels x 4 k-chunks; ISA 16-bit A layout) ----------
  v16h Af[NCHAN][4];
  {
    const _Float16* lp = chan + (lane & 15) * CHAN_ROW_H + ((lane >> 4) ? 8 : 0);
    #pragma unroll
    for (int c = 0; c < NCHAN; ++c) {
      #pragma unroll
      for (int kc = 0; kc < 4; ++kc) {
        const _Float16* p = lp + c * CHAN_H + kc * 32;
        union { uint4 u[2]; v16h h; } r;
        r.u[0] = *(const uint4*)p;          // K = base+{0..7}   (hi half-wave: +8)
        r.u[1] = *(const uint4*)(p + 16);   // K = base+{16..23} (hi half-wave: +24)
        Af[c][kc] = r.h;
      }
    }
  }

  // ---------- layer 2 (WMMA f16, K=128), tanh jet 2, layer 3 dot, reduce ----------
  float lacc[8] = {0, 0, 0, 0, 0, 0, 0, 0};
  const int ncol = lane & 15;

  #pragma unroll
  for (int n = 0; n < 4; ++n) {
    v16h Bn[4];
    #pragma unroll
    for (int kc = 0; kc < 4; ++kc) {
      int fk = n * 4 + kc;
      union { uint4 u[2]; v16h h; } r;
      r.u[0] = bfrag4[(fk * 2 + 0) * 32 + lane];
      r.u[1] = bfrag4[(fk * 2 + 1) * 32 + lane];
      Bn[kc] = r.h;
    }
    v8f acc[NCHAN] = {};
    #pragma unroll
    for (int kc = 0; kc < 4; ++kc) {
      #pragma unroll
      for (int c = 0; c < NCHAN; ++c)
        acc[c] = __builtin_amdgcn_wmma_f32_16x16x32_f16(
            false, Af[c][kc], false, Bn[kc], (short)0, acc[c], false, false);
    }

    float b2v = b2s[n * 16 + ncol];
    float w3v = w3s[n * 16 + ncol];
    #pragma unroll
    for (int r = 0; r < 8; ++r) {
      float tt  = fast_tanh(acc[0][r] + b2v);
      float s2  = 1.0f - tt * tt;
      float z2s = acc[4][r];                                // Σ_d z2_d (pre-summed)
      float z1q = acc[1][r] * acc[1][r] + acc[2][r] * acc[2][r] + acc[3][r] * acc[3][r];
      // Σ_d [ s2*z2_d - 2*t*s2*z1_d^2 ] * W3[N]
      lacc[r] = fmaf(w3v * s2, z2s - 2.0f * tt * z1q, lacc[r]);
    }
  }

  // cross-lane reduce over N (16 lanes per half-wave hold N=0..15 of this M-group)
  #pragma unroll
  for (int off = 8; off >= 1; off >>= 1) {
    #pragma unroll
    for (int r = 0; r < 8; ++r)
      lacc[r] += __shfl_xor(lacc[r], off, 16);
  }

  if ((lane & 15) == 0) {
    int base = tileBase + (lane >> 4) * 8;   // lanes 0..15 -> M=r ; 16..31 -> M=8+r
    #pragma unroll
    for (int r = 0; r < 8; ++r)
      if (base + r < B) out[base + r] = lacc[r];
  }
}

extern "C" void kernel_launch(void* const* d_in, const int* in_sizes, int n_in,
                              void* d_out, int out_size, void* d_ws, size_t ws_size,
                              hipStream_t stream) {
  const float* x  = (const float*)d_in[0];
  const float* W1 = (const float*)d_in[1];
  const float* b1 = (const float*)d_in[2];
  const float* W2 = (const float*)d_in[3];
  const float* b2 = (const float*)d_in[4];
  const float* W3 = (const float*)d_in[5];
  // d_in[6] (b3) only shifts the 0th Taylor coefficient; the Laplacian ignores it.
  float* out = (float*)d_out;

  int B = in_sizes[0] / 3;
  int rowsPerBlock = WAVES_PER_BLOCK * ROWS_PER_WAVE;  // 64
  int blocks = (B + rowsPerBlock - 1) / rowsPerBlock;

  lap_mlp_kernel<<<blocks, WAVES_PER_BLOCK * 32, SMEM_BYTES, stream>>>(
      x, W1, b1, W2, b2, W3, out, B);
}
